// SICNet_72395968741395
// MI455X (gfx1250) — matrix operand order repaired and checked by reading, not compiled, not run on hardware
//
#include <hip/hip_runtime.h>
#include <hip/hip_bf16.h>
#include <math.h>

// ---------------------------------------------------------------------------
// SICNet on MI455X (gfx1250): all matmuls via v_wmma_f32_16x16x32_bf16.
//  - k_gemm_xp : batched input projections, A-tile staged in LDS via
//                GLOBAL_LOAD_ASYNC_TO_LDS (ASYNCcnt)
//  - k_gru_rec : persistent recurrence, Whh in VGPR B-fragments, h in LDS,
//                xp tile double-buffered in LDS with async prefetch
//  - small kernels: embed gather, weight f32->bf16, transpose, head
// ---------------------------------------------------------------------------

#define B_   256
#define S_   512
#define F_   128
#define H3_  384
#define T2_  256
#define K2_  512

typedef __attribute__((ext_vector_type(16))) __bf16 v16bf;
typedef __attribute__((ext_vector_type(8)))  float  v8f;
typedef int v4i_ __attribute__((vector_size(16)));   // builtin's pointee type

__device__ __forceinline__ float sigmoidf_(float x) { return 1.f / (1.f + __expf(-x)); }

// ---------------- async global->LDS copy (ASYNCcnt path) -------------------
#if __has_builtin(__builtin_amdgcn_global_load_async_to_lds_b128)
#define HAS_ASYNC_LDS 1
#else
#define HAS_ASYNC_LDS 0
#endif

__device__ __forceinline__ void async_b128(const __bf16* g, __bf16* l) {
#if HAS_ASYNC_LDS
  __builtin_amdgcn_global_load_async_to_lds_b128(
      (__attribute__((address_space(1))) v4i_*)g,
      (__attribute__((address_space(3))) v4i_*)l, 0, 0);
#else
  *reinterpret_cast<uint4*>(l) = *reinterpret_cast<const uint4*>(g);
#endif
}

#if HAS_ASYNC_LDS
#if __has_builtin(__builtin_amdgcn_s_wait_asynccnt)
#define WAIT_ASYNC(n) __builtin_amdgcn_s_wait_asynccnt(n)
#else
#define WAIT_ASYNC(n) asm volatile("s_wait_asynccnt %0" ::"i"(n) : "memory")
#endif
#else
#define WAIT_ASYNC(n) ((void)0)
#endif

// A-matrix 16x32 bf16 fragment: lane(&15)=M row; lanes 0-15 hold K 0..7 & 16..23,
// lanes 16-31 hold K 8..15 & 24..31 (ISA 7.12.2).
__device__ __forceinline__ v16bf load_a_frag(const __bf16* row, int kbase, int hi) {
  v16bf f;
  const __bf16* p0 = row + kbase + hi * 8;
  const __bf16* p1 = row + kbase + 16 + hi * 8;
#pragma unroll
  for (int i = 0; i < 8; ++i) { f[i] = p0[i]; f[8 + i] = p1[i]; }
  return f;
}

// B-matrix 32x16 bf16 fragment: lane(&15)=N col (== weight row, since B=W^T);
// lanes 0-15 hold K 0..7 & 16..23, lanes 16-31 hold K 8..15 & 24..31.
__device__ __forceinline__ v16bf load_b_frag(const __bf16* wrow, int kbase, int hi) {
  v16bf f;
  const __bf16* p0 = wrow + kbase + hi * 8;
  const __bf16* p1 = wrow + kbase + 16 + hi * 8;
#pragma unroll
  for (int i = 0; i < 8; ++i) { f[i] = p0[i]; f[8 + i] = p1[i]; }
  return f;
}

__device__ __forceinline__ v8f wmma_bf16(v16bf a, v16bf b, v8f c) {
  return __builtin_amdgcn_wmma_f32_16x16x32_bf16(false, a, false, b, (short)0, c,
                                                 false, false);
}

// --------------------------- utility kernels -------------------------------

__global__ void k_f2bf(const float* __restrict__ src, __bf16* __restrict__ dst, int n) {
  int i = blockIdx.x * blockDim.x + threadIdx.x;
  if (i < n) dst[i] = (__bf16)src[i];
}

// e[(b*S+s)*128 + c] = bf16(embed[x[b*S+s]][c])
__global__ void k_embed(const int* __restrict__ x, const float* __restrict__ embed,
                        __bf16* __restrict__ e) {
  int idx = blockIdx.x * blockDim.x + threadIdx.x;   // over B*S*128
  int c = idx & 127;
  int r = idx >> 7;
  e[idx] = (__bf16)embed[(size_t)x[r] * F_ + c];
}

// h2t[(b*256+f)*512 + s] = act1[(b*512+s)*256 + f]
__global__ void k_transpose(const __bf16* __restrict__ in, __bf16* __restrict__ out) {
  int idx = blockIdx.x * blockDim.x + threadIdx.x;   // over B*256*512
  int s = idx & 511;
  int f = (idx >> 9) & 255;
  int b = idx >> 17;
  out[idx] = in[((size_t)b * S_ + s) * 256 + f];
}

// ------------------------ batched input projection -------------------------
// out[m, n] = sum_k A[m,k] * W[n,k] + bih[n]   (M x 384, bf16 out, f32 accum)
// grid.x = M/16; block = 256 (8 waves). A-tile (16 x K) staged once in LDS via
// async copies; wave w owns N-tiles {w, w+8, w+16} == the r/z/n gate columns.
__global__ void __launch_bounds__(256)
k_gemm_xp(const __bf16* __restrict__ A, const __bf16* __restrict__ W,
          const float* __restrict__ bih, __bf16* __restrict__ out, int K) {
  __shared__ __bf16 ash[16 * K2_];

  const int m0   = blockIdx.x * 16;
  const int wave = threadIdx.x >> 5;
  const int lane = threadIdx.x & 31;
  const int n    = lane & 15;
  const int hi   = lane >> 4;

  // stage A-tile: 16 rows x K bf16 = 2K chunks of 16B
  const int cpr = K >> 3;                 // 16B chunks per row (pow2)
  const int sh  = 31 - __clz(cpr);
  const int nch = cpr * 16;
  for (int c = threadIdx.x; c < nch; c += 256) {
    const int row = c >> sh;
    const int cc  = c & (cpr - 1);
    async_b128(A + (size_t)(m0 + row) * K + cc * 8, ash + row * K + cc * 8);
  }
  WAIT_ASYNC(0);
  __syncthreads();

  const __bf16* arow = ash + n * K;
  v8f acc[3] = {v8f{}, v8f{}, v8f{}};

  for (int kb = 0; kb < K; kb += 32) {
    v16bf a = load_a_frag(arow, kb, hi);
#pragma unroll
    for (int g = 0; g < 3; ++g) {
      const int col = (wave + g * 8) * 16 + n;           // 0..383
      v16bf b = load_b_frag(W + (size_t)col * K, kb, hi);
      acc[g] = wmma_bf16(a, b, acc[g]);
    }
  }
#pragma unroll
  for (int g = 0; g < 3; ++g) {
    const int col  = (wave + g * 8) * 16 + n;
    const float bv = bih[col];
#pragma unroll
    for (int v = 0; v < 8; ++v) {
      const int m = m0 + v + hi * 8;
      out[(size_t)m * H3_ + col] = (__bf16)(acc[g][v] + bv);
    }
  }
}

// ----------------------------- GRU recurrence ------------------------------
// One block = 16 batch rows for the full sequence. 8 waves x 16 hidden cols.
// Whh B-fragments (3 gates x 4 K-frags = 96 VGPRs/lane) pre-loaded once.
// h (16x128 bf16) lives in LDS; xp tile (16x384 bf16) double-buffered in LDS
// with async prefetch of step t+1 overlapping the WMMAs of step t.
__global__ void __launch_bounds__(256)
k_gru_rec(const __bf16* __restrict__ xp,    // (NB*T) x 384, bih folded in
          const __bf16* __restrict__ whh,   // 384 x 128 bf16
          const float* __restrict__ bhh,    // 384
          int T, int reverse,
          __bf16* __restrict__ out, int out_stride, int out_coff) {
  __shared__ __bf16 hsh[16 * 128];
  __shared__ __bf16 xsh[2][16 * H3_];

  const int b0   = blockIdx.x * 16;
  const int wave = threadIdx.x >> 5;
  const int lane = threadIdx.x & 31;
  const int n    = lane & 15;
  const int hi   = lane >> 4;
  const int j    = wave * 16 + n;            // hidden column owned by this lane

  // Pre-load recurrent weight fragments into VGPRs (stay resident all steps).
  v16bf bfr[3][4];
#pragma unroll
  for (int g = 0; g < 3; ++g)
#pragma unroll
    for (int kf = 0; kf < 4; ++kf)
      bfr[g][kf] = load_b_frag(whh + (size_t)(g * 128 + j) * 128, kf * 32, hi);

  const float bh0 = bhh[j], bh1 = bhh[128 + j], bh2 = bhh[256 + j];

  // issue async copies of the (16 x 384) xp tile at time tstep into buffer buf
  auto issue_xp = [&](int tstep, int buf) {
    for (int c = threadIdx.x; c < 768; c += 256) {   // 3 chunks per thread
      const int row = c / 48;
      const int cc  = c - row * 48;
      async_b128(xp + ((size_t)(b0 + row) * T + tstep) * H3_ + cc * 8,
                 &xsh[buf][row * H3_ + cc * 8]);
    }
  };

  issue_xp(reverse ? (T - 1) : 0, 0);                // preload step 0

  for (int i = threadIdx.x; i < 16 * 128; i += blockDim.x) hsh[i] = (__bf16)0.f;
  __syncthreads();

  for (int t = 0; t < T; ++t) {
    const int tt = reverse ? (T - 1 - t) : t;
    const int tn = (t + 1 < T) ? (reverse ? (T - 2 - t) : (t + 1)) : tt;

    issue_xp(tn, (t + 1) & 1);                       // prefetch next step

    // gates_hh = h @ Whh^T  (M=16, N=48 per wave, K=128)
    v8f acc[3] = {v8f{}, v8f{}, v8f{}};
#pragma unroll
    for (int kf = 0; kf < 4; ++kf) {
      v16bf a = load_a_frag(hsh + n * 128, kf * 32, hi);
#pragma unroll
      for (int g = 0; g < 3; ++g) acc[g] = wmma_bf16(a, bfr[g][kf], acc[g]);
    }

    WAIT_ASYNC(3);         // own older (step t) tile complete; t+1 stays in flight
    __syncthreads();       // cross-thread visibility of this step's xp tile

    const __bf16* xt = xsh[t & 1];
    float hnew[8];
#pragma unroll
    for (int v = 0; v < 8; ++v) {
      const int m = v + hi * 8;
      const __bf16* xr = xt + m * H3_;
      const float xv_r = (float)xr[j];
      const float xv_z = (float)xr[128 + j];
      const float xv_n = (float)xr[256 + j];
      const float hr = acc[0][v] + bh0;
      const float hz = acc[1][v] + bh1;
      const float hn = acc[2][v] + bh2;          // r multiplies (hh_n + bhh_n)
      const float r  = sigmoidf_(xv_r + hr);
      const float z  = sigmoidf_(xv_z + hz);
      const float nn = tanhf(xv_n + r * hn);
      const float ho = (float)hsh[m * 128 + j];
      hnew[v] = (1.f - z) * nn + z * ho;
    }
    __syncthreads();                              // all reads of h done
#pragma unroll
    for (int v = 0; v < 8; ++v) {
      const int m = v + hi * 8;
      hsh[m * 128 + j] = (__bf16)hnew[v];
      const size_t row = (size_t)(b0 + m) * T + tt;
      out[row * (size_t)out_stride + out_coff + j] = (__bf16)hnew[v];
    }
    __syncthreads();                              // h published for next step
  }
}

// --------------------------------- head ------------------------------------
// per batch: dd = sigmoid(w21 . h2[:,k]) * tanh(w22 . h2[:,k]); y = leaky(dd@fc1^T+b)
__global__ void k_head1(const __bf16* __restrict__ h2,  // (B*256) x 128
                        const float* __restrict__ w21, const float* __restrict__ w22,
                        const float* __restrict__ fc1_w, const float* __restrict__ fc1_b,
                        float* __restrict__ y) {
  __shared__ float sdd[128];
  const int b = blockIdx.x;
  const int k = threadIdx.x;     // 128 threads
  const __bf16* base = h2 + (size_t)b * 256 * 128 + k;
  float a1 = 0.f, a2 = 0.f;
  for (int f = 0; f < 256; ++f) {
    const float v = (float)base[(size_t)f * 128];
    a1 += w21[f] * v;
    a2 += w22[f] * v;
  }
  sdd[k] = sigmoidf_(a1) * tanhf(a2);
  __syncthreads();
  if (k < 16) {
    float acc = fc1_b[k];
    for (int c = 0; c < 128; ++c) acc += sdd[c] * fc1_w[k * 128 + c];
    y[b * 16 + k] = acc > 0.f ? acc : 0.2f * acc;
  }
}

// batchnorm (batch stats) + fc2 + log_softmax; one block of 256 threads.
__global__ void k_head2(const float* __restrict__ y, const float* __restrict__ gamma,
                        const float* __restrict__ beta, const float* __restrict__ fc2_w,
                        const float* __restrict__ fc2_b, float* __restrict__ out) {
  __shared__ float ssum[16], ssq[16];
  if (threadIdx.x < 16) { ssum[threadIdx.x] = 0.f; ssq[threadIdx.x] = 0.f; }
  __syncthreads();
  const int b = threadIdx.x;
  float yv[16];
#pragma unroll
  for (int k = 0; k < 16; ++k) {
    yv[k] = y[b * 16 + k];
    atomicAdd(&ssum[k], yv[k]);
    atomicAdd(&ssq[k], yv[k] * yv[k]);
  }
  __syncthreads();
  float z0 = fc2_b[0], z1 = fc2_b[1];
#pragma unroll
  for (int k = 0; k < 16; ++k) {
    const float mu  = ssum[k] * (1.f / 256.f);
    const float var = ssq[k] * (1.f / 256.f) - mu * mu;
    const float tn  = (yv[k] - mu) * rsqrtf(var + 1e-5f) * gamma[k] + beta[k];
    z0 += fc2_w[k] * tn;
    z1 += fc2_w[16 + k] * tn;
  }
  const float mx  = fmaxf(z0, z1);
  const float lse = mx + __logf(__expf(z0 - mx) + __expf(z1 - mx));
  out[b * 2 + 0] = z0 - lse;
  out[b * 2 + 1] = z1 - lse;
}

// ------------------------------- launcher ----------------------------------
// Input order (setup_inputs insertion order, gru dicts flattened in order):
//  0:x 1:embed
//  2..5  : gru1.l0f Wih,Whh,bih,bhh     6..9  : gru1.l0b
// 10..13 : gru1.l1f                    14..17 : gru1.l1b
// 18..21 : gru2.l0                     22..25 : gru2.l1
// 26:w21 27:w22 28:fc1_w 29:fc1_b 30:bn_gamma 31:bn_beta 32:fc2_w 33:fc2_b
extern "C" void kernel_launch(void* const* d_in, const int* in_sizes, int n_in,
                              void* d_out, int out_size, void* d_ws, size_t ws_size,
                              hipStream_t stream) {
  (void)in_sizes; (void)n_in; (void)out_size; (void)ws_size;
  char* W = (char*)d_ws;
  size_t off = 0;
  auto nxt = [&](size_t bytes) {
    size_t r = off;
    off += (bytes + 255) & ~(size_t)255;
    return r;
  };
  const size_t o_xp   = nxt((size_t)131072 * 384 * 2);   // projection scratch (reused)
  const size_t o_e    = nxt((size_t)131072 * 128 * 2);   // embed; later g2a/g2b
  const size_t o_act0 = nxt((size_t)131072 * 256 * 2);   // gru1 l0 out; later h2t
  const size_t o_act1 = nxt((size_t)131072 * 256 * 2);   // gru1 l1 out
  // bf16 weights: {src input idx, elems}
  const int widx[12] = {2, 3, 6, 7, 10, 11, 14, 15, 18, 19, 22, 23};
  const int wcnt[12] = {49152, 49152, 49152, 49152, 98304, 49152,
                        98304, 49152, 196608, 49152, 49152, 49152};
  size_t o_w[12];
  for (int i = 0; i < 12; ++i) o_w[i] = nxt((size_t)wcnt[i] * 2);
  const size_t o_y = nxt((size_t)B_ * 16 * 4);

  __bf16* xp   = (__bf16*)(W + o_xp);
  __bf16* e    = (__bf16*)(W + o_e);
  __bf16* act0 = (__bf16*)(W + o_act0);
  __bf16* act1 = (__bf16*)(W + o_act1);
  __bf16* h2t  = act0;                                     // reuse
  __bf16* g2a  = (__bf16*)(W + o_e);                       // reuse
  __bf16* g2b  = (__bf16*)(W + o_e + (size_t)65536 * 128 * 2);
  float*  yb   = (float*)(W + o_y);
  __bf16* wb[12];
  for (int i = 0; i < 12; ++i) wb[i] = (__bf16*)(W + o_w[i]);

  // 1) weights -> bf16
  for (int i = 0; i < 12; ++i)
    k_f2bf<<<(wcnt[i] + 255) / 256, 256, 0, stream>>>((const float*)d_in[widx[i]],
                                                      wb[i], wcnt[i]);
  // 2) embedding gather
  k_embed<<<(B_ * S_ * F_) / 256, 256, 0, stream>>>((const int*)d_in[0],
                                                    (const float*)d_in[1], e);

  const int M1 = B_ * S_;   // 131072
  const int M2 = B_ * T2_;  // 65536

  // 3) gru1 layer0 forward / backward
  k_gemm_xp<<<M1 / 16, 256, 0, stream>>>(e, wb[0], (const float*)d_in[4], xp, 128);
  k_gru_rec<<<16, 256, 0, stream>>>(xp, wb[1], (const float*)d_in[5], S_, 0, act0, 256, 0);
  k_gemm_xp<<<M1 / 16, 256, 0, stream>>>(e, wb[2], (const float*)d_in[8], xp, 128);
  k_gru_rec<<<16, 256, 0, stream>>>(xp, wb[3], (const float*)d_in[9], S_, 1, act0, 256, 128);

  // 4) gru1 layer1 forward / backward (input = act0, K=256)
  k_gemm_xp<<<M1 / 16, 256, 0, stream>>>(act0, wb[4], (const float*)d_in[12], xp, 256);
  k_gru_rec<<<16, 256, 0, stream>>>(xp, wb[5], (const float*)d_in[13], S_, 0, act1, 256, 0);
  k_gemm_xp<<<M1 / 16, 256, 0, stream>>>(act0, wb[6], (const float*)d_in[16], xp, 256);
  k_gru_rec<<<16, 256, 0, stream>>>(xp, wb[7], (const float*)d_in[17], S_, 1, act1, 256, 128);

  // 5) transpose (B,S,2F) -> (B,2F,S)
  k_transpose<<<(B_ * 256 * S_) / 256, 256, 0, stream>>>(act1, h2t);

  // 6) gru2 layer0 (K=512, T=256) and layer1 (K=128, T=256)
  k_gemm_xp<<<M2 / 16, 256, 0, stream>>>(h2t, wb[8], (const float*)d_in[20], xp, 512);
  k_gru_rec<<<16, 256, 0, stream>>>(xp, wb[9], (const float*)d_in[21], T2_, 0, g2a, 128, 0);
  k_gemm_xp<<<M2 / 16, 256, 0, stream>>>(g2a, wb[10], (const float*)d_in[24], xp, 128);
  k_gru_rec<<<16, 256, 0, stream>>>(xp, wb[11], (const float*)d_in[25], T2_, 0, g2b, 128, 0);

  // 7) head
  k_head1<<<B_, 128, 0, stream>>>(g2b, (const float*)d_in[26], (const float*)d_in[27],
                                  (const float*)d_in[28], (const float*)d_in[29], yb);
  k_head2<<<1, 256, 0, stream>>>(yb, (const float*)d_in[30], (const float*)d_in[31],
                                 (const float*)d_in[32], (const float*)d_in[33],
                                 (float*)d_out);
}
